// SimplifiedVulnerabilityGNN_30923764531876
// MI455X (gfx1250) — compile-verified, matrix operand order checked by reading, not run on hardware
//
#include <hip/hip_runtime.h>

typedef __bf16 bf16;
typedef __attribute__((ext_vector_type(16))) __bf16 v16bf;
typedef __attribute__((ext_vector_type(8)))  __bf16 bf16x8;
typedef __attribute__((ext_vector_type(8)))  float  v8f;
typedef __attribute__((address_space(3)))    bf16   lds_bf16;

#define TM  64
#define TNB 128
#define TK  32

// Flip to 0 to fall back to synchronous staging.
#define USE_ASYNC_LDS 1

union V16 { v16bf v; bf16x8 h[2]; };

// ---------------------------------------------------------------------------
// bf16 WMMA GEMM: C[M,N] = act(A[M,K] @ Bt[N,K]^T + bias), fp32 accum.
// A row-major [M,K]; B pre-transposed as Bt [N,K] so both LDS tiles are
// K-major and every WMMA fragment is two contiguous ds_load_b128.
// Block = 256 threads (8 waves), tile 64x128, K-step 32, double-buffered
// LDS fed by global_load_async_to_lds_b128 (ASYNCcnt), 1 barrier/step.
// Wave w: rows (w&3)*16, cols (w>>2)*64 -> four 16x16 WMMA subtiles,
// A fragment reused across the 4 WMMAs; all B fragments loaded up front so
// the WMMAs issue back-to-back without per-WMMA dscnt stalls.
// Requires N % 128 == 0, K % 32 == 0 (true here); M ragged.
// ---------------------------------------------------------------------------
__global__ __launch_bounds__(256) void wmma_gemm_bf16(
    const bf16* __restrict__ A, const bf16* __restrict__ Bt,
    const float* __restrict__ bias, float* __restrict__ C,
    bf16* __restrict__ Cbf, int M, int N, int K, int act) {
  __shared__ bf16 As[2][TM][TK];
  __shared__ bf16 Bs[2][TNB][TK];

  const int m0   = blockIdx.y * TM;
  const int n0   = blockIdx.x * TNB;
  const int tid  = threadIdx.x;
  const int w    = tid >> 5;
  const int lane = tid & 31;
  const int mw   = (w & 3) * 16;
  const int nw   = (w >> 2) * 64;
  const int lhalf = lane >> 4;
  const int l16   = lane & 15;

  // staging map: A = 64 rows x 32 K (1 b128/thread), B = 128 rows (2 b128/thread)
  const int srow = tid >> 2;
  const int scol = (tid & 3) * 8;
  const int gm   = m0 + srow;                  // A source row (may exceed M)
  const size_t aoff  = (size_t)gm * K + scol;                  // + k0 per step
  const size_t boff0 = (size_t)(n0 + srow) * K + scol;
  const size_t boff1 = (size_t)(n0 + srow + 64) * K + scol;

#if USE_ASYNC_LDS
  const unsigned ldsA0  = (unsigned)(unsigned long long)(lds_bf16*)&As[0][srow][scol];
  const unsigned ldsA1  = (unsigned)(unsigned long long)(lds_bf16*)&As[1][srow][scol];
  const unsigned ldsB00 = (unsigned)(unsigned long long)(lds_bf16*)&Bs[0][srow][scol];
  const unsigned ldsB01 = (unsigned)(unsigned long long)(lds_bf16*)&Bs[1][srow][scol];
  const unsigned ldsB10 = (unsigned)(unsigned long long)(lds_bf16*)&Bs[0][srow + 64][scol];
  const unsigned ldsB11 = (unsigned)(unsigned long long)(lds_bf16*)&Bs[1][srow + 64][scol];

  auto issue_tile = [&](int k0, int q) {
    if (gm < M) {  // out-of-range A rows left stale: they only feed unstored D rows
      unsigned long long ga = (unsigned long long)(A + aoff + k0);
      asm volatile("global_load_async_to_lds_b128 %0, %1, off"
                   :: "v"(q ? ldsA1 : ldsA0), "v"(ga) : "memory");
    }
    unsigned long long gb0 = (unsigned long long)(Bt + boff0 + k0);
    unsigned long long gb1 = (unsigned long long)(Bt + boff1 + k0);
    asm volatile("global_load_async_to_lds_b128 %0, %1, off"
                 :: "v"(q ? ldsB01 : ldsB00), "v"(gb0) : "memory");
    asm volatile("global_load_async_to_lds_b128 %0, %1, off"
                 :: "v"(q ? ldsB11 : ldsB10), "v"(gb1) : "memory");
  };
#endif

  const int ar = mw + l16;        // fragment rows in LDS
  const int kb = lhalf * 8;       // per-lane K base within 32-wide step

  v8f acc[4] = {{}, {}, {}, {}};

  const int nsteps = K / TK;
#if USE_ASYNC_LDS
  issue_tile(0, 0);               // prologue: tile 0 -> buffer 0
  for (int step = 0; step < nsteps; ++step) {
    const int q = step & 1;
    asm volatile("s_wait_asynccnt 0x0" ::: "memory");  // tile `step` landed
    __syncthreads();                                   // visible to all waves
    if (step + 1 < nsteps) issue_tile((step + 1) * TK, q ^ 1);
#else
  for (int step = 0; step < nsteps; ++step) {
    const int q = step & 1;
    {
      const int k0 = step * TK;
      if (gm < M) *(bf16x8*)(&As[q][srow][scol]) = *(const bf16x8*)(A + aoff + k0);
      *(bf16x8*)(&Bs[q][srow][scol])      = *(const bf16x8*)(Bt + boff0 + k0);
      *(bf16x8*)(&Bs[q][srow + 64][scol]) = *(const bf16x8*)(Bt + boff1 + k0);
    }
    __syncthreads();
#endif

    // Load A fragment + ALL four B fragments into distinct registers first,
    // then issue the 4 WMMAs back-to-back (shared A, disjoint accumulators:
    // no D->A/B hazard, and partial dscnt waits can overlap load latency).
    V16 a, b0, b1, b2, b3;
    a.h[0]  = *(const bf16x8*)(&As[q][ar][kb]);
    a.h[1]  = *(const bf16x8*)(&As[q][ar][16 + kb]);
    b0.h[0] = *(const bf16x8*)(&Bs[q][nw +  0 + l16][kb]);
    b0.h[1] = *(const bf16x8*)(&Bs[q][nw +  0 + l16][16 + kb]);
    b1.h[0] = *(const bf16x8*)(&Bs[q][nw + 16 + l16][kb]);
    b1.h[1] = *(const bf16x8*)(&Bs[q][nw + 16 + l16][16 + kb]);
    b2.h[0] = *(const bf16x8*)(&Bs[q][nw + 32 + l16][kb]);
    b2.h[1] = *(const bf16x8*)(&Bs[q][nw + 32 + l16][16 + kb]);
    b3.h[0] = *(const bf16x8*)(&Bs[q][nw + 48 + l16][kb]);
    b3.h[1] = *(const bf16x8*)(&Bs[q][nw + 48 + l16][16 + kb]);

    acc[0] = __builtin_amdgcn_wmma_f32_16x16x32_bf16(false, a.v, false, b0.v, (short)0, acc[0], false, false);
    acc[1] = __builtin_amdgcn_wmma_f32_16x16x32_bf16(false, a.v, false, b1.v, (short)0, acc[1], false, false);
    acc[2] = __builtin_amdgcn_wmma_f32_16x16x32_bf16(false, a.v, false, b2.v, (short)0, acc[2], false, false);
    acc[3] = __builtin_amdgcn_wmma_f32_16x16x32_bf16(false, a.v, false, b3.v, (short)0, acc[3], false, false);

#if !USE_ASYNC_LDS
    __syncthreads();
#endif
  }

#pragma unroll
  for (int r = 0; r < 8; ++r) {
    int gmo = m0 + mw + r + (lhalf << 3);
    if (gmo >= M) continue;
#pragma unroll
    for (int j = 0; j < 4; ++j) {
      int gn = n0 + nw + j * 16 + l16;
      float v = acc[j][r] + (bias ? bias[gn] : 0.f);
      if (act == 1) v = fmaxf(v, 0.f);
      size_t idx = (size_t)gmo * N + gn;
      C[idx] = v;
      if (Cbf) Cbf[idx] = (bf16)v;
    }
  }
}

// ---------------------------------------------------------------------------
// Elementwise helpers
// ---------------------------------------------------------------------------
__global__ void cvt_bf16_kernel(const float* __restrict__ in, bf16* __restrict__ out, size_t n) {
  size_t i = (size_t)blockIdx.x * blockDim.x + threadIdx.x;
  if (i < n) out[i] = (bf16)in[i];
}

// in: f32 [K][N] row-major -> out: bf16 [N][K] (transposed, K-major)
__global__ void cvt_tr_bf16_kernel(const float* __restrict__ in, bf16* __restrict__ out,
                                   int K, int N) {
  size_t i = (size_t)blockIdx.x * blockDim.x + threadIdx.x;
  if (i >= (size_t)K * N) return;
  int n = (int)(i / K), k = (int)(i % K);
  out[i] = (bf16)in[(size_t)k * N + n];
}

__global__ void fill_f32_kernel(float* __restrict__ p, float v, size_t n) {
  size_t i = (size_t)blockIdx.x * blockDim.x + threadIdx.x;
  if (i < n) p[i] = v;
}

// h = elu(h + bias[c]); optional bf16 copy
__global__ void bias_elu_kernel(float* __restrict__ h, const float* __restrict__ bias,
                                bf16* __restrict__ hbf, size_t total, int C) {
  size_t i = (size_t)blockIdx.x * blockDim.x + threadIdx.x;
  if (i >= total) return;
  float v = h[i] + bias[(int)(i % (size_t)C)];
  v = (v > 0.f) ? v : expm1f(v);
  h[i] = v;
  if (hbf) hbf[i] = (bf16)v;
}

// ---------------------------------------------------------------------------
// Attention scores: one wave32 per (node, head). a_s/a_d: [N,4]
// ---------------------------------------------------------------------------
__global__ void scores_kernel(const float* __restrict__ hW,
                              const float* __restrict__ att_s, const float* __restrict__ att_d,
                              float* __restrict__ a_s, float* __restrict__ a_d, int N) {
  int wid  = (int)(((size_t)blockIdx.x * blockDim.x + threadIdx.x) >> 5);
  int lane = threadIdx.x & 31;
  int n = wid >> 2, h = wid & 3;
  if (n >= N) return;
  const float* row = hW + (size_t)n * 1024 + h * 256;
  const float* as  = att_s + h * 256;
  const float* ad  = att_d + h * 256;
  float ss = 0.f, sd = 0.f;
  for (int i = lane; i < 256; i += 32) {
    float v = row[i];
    ss += v * as[i];
    sd += v * ad[i];
  }
#pragma unroll
  for (int off = 16; off > 0; off >>= 1) {
    ss += __shfl_xor(ss, off, 32);
    sd += __shfl_xor(sd, off, 32);
  }
  if (lane == 0) { a_s[n * 4 + h] = ss; a_d[n * 4 + h] = sd; }
}

// ---------------------------------------------------------------------------
// Edge kernels (E real edges + N self loops)
// ---------------------------------------------------------------------------
__device__ __forceinline__ void edge_nodes(const int* __restrict__ ei, int E, int e, int& s, int& d) {
  if (e < E) { s = ei[e]; d = ei[E + e]; } else { s = e - E; d = e - E; }
}

__device__ __forceinline__ float lrelu02(float v) { return v > 0.f ? v : 0.2f * v; }

// order-preserving float atomic max (init must be very negative)
__device__ __forceinline__ void atomicMaxF(float* addr, float val) {
  if (val >= 0.f) atomicMax((int*)addr, __float_as_int(val));
  else            atomicMin((unsigned int*)addr, __float_as_uint(val));
}

__global__ void edge_max_kernel(const int* __restrict__ ei, int E, int N,
                                const float* __restrict__ a_s, const float* __restrict__ a_d,
                                float* __restrict__ mbuf) {
  size_t t = (size_t)blockIdx.x * blockDim.x + threadIdx.x;
  int E2 = E + N;
  if (t >= (size_t)E2 * 4) return;
  int e = (int)(t >> 2), h = (int)(t & 3);
  int s, d; edge_nodes(ei, E, e, s, d);
  float v = lrelu02(a_s[s * 4 + h] + a_d[d * 4 + h]);
  atomicMaxF(&mbuf[d * 4 + h], v);
}

__global__ void edge_sum_kernel(const int* __restrict__ ei, int E, int N,
                                const float* __restrict__ a_s, const float* __restrict__ a_d,
                                const float* __restrict__ mbuf, float* __restrict__ dbuf) {
  size_t t = (size_t)blockIdx.x * blockDim.x + threadIdx.x;
  int E2 = E + N;
  if (t >= (size_t)E2 * 4) return;
  int e = (int)(t >> 2), h = (int)(t & 3);
  int s, d; edge_nodes(ei, E, e, s, d);
  float v = lrelu02(a_s[s * 4 + h] + a_d[d * 4 + h]);
  atomicAdd(&dbuf[d * 4 + h], expf(v - mbuf[d * 4 + h]));
}

// one wave32 per edge; lane strides the 1024 features; alpha recomputed on the fly
__global__ void edge_agg_kernel(const int* __restrict__ ei, int E, int N,
                                const float* __restrict__ a_s, const float* __restrict__ a_d,
                                const float* __restrict__ mbuf, const float* __restrict__ dbuf,
                                const float* __restrict__ hW, float* __restrict__ agg) {
  int wid  = (int)(((size_t)blockIdx.x * blockDim.x + threadIdx.x) >> 5);
  int lane = threadIdx.x & 31;
  int E2 = E + N;
  if (wid >= E2) return;
  int s, d; edge_nodes(ei, E, wid, s, d);
  float alpha[4];
#pragma unroll
  for (int h = 0; h < 4; ++h) {
    float v = lrelu02(a_s[s * 4 + h] + a_d[d * 4 + h]);
    alpha[h] = expf(v - mbuf[d * 4 + h]) / (dbuf[d * 4 + h] + 1e-16f);
  }
  const float* srcr = hW + (size_t)s * 1024;
  float* dstr = agg + (size_t)d * 1024;
  for (int c = lane; c < 1024; c += 32)
    atomicAdd(&dstr[c], srcr[c] * alpha[c >> 8]);
}

// ---------------------------------------------------------------------------
// Pooling
// ---------------------------------------------------------------------------
__global__ void pool_accum_kernel(const float* __restrict__ h, const int* __restrict__ batch,
                                  float* __restrict__ meansum, float* __restrict__ maxp,
                                  int* __restrict__ cnt, int N) {
  size_t i = (size_t)blockIdx.x * blockDim.x + threadIdx.x;
  if (i >= (size_t)N * 1024) return;
  int n = (int)(i >> 10);
  int c = (int)(i & 1023);
  int g = batch[n];
  float v = h[i];
  atomicAdd(&meansum[g * 1024 + c], v);
  atomicMaxF(&maxp[g * 1024 + c], v);
  if (c == 0) atomicAdd(&cnt[g], 1);
}

__global__ void pool_finalize_kernel(const float* __restrict__ meansum, const float* __restrict__ maxp,
                                     const int* __restrict__ cnt, float* __restrict__ g,
                                     bf16* __restrict__ gbf, int G) {
  int i = blockIdx.x * blockDim.x + threadIdx.x;
  if (i >= G * 2048) return;
  int gr = i >> 11, c = i & 2047;
  float v = (c < 1024) ? meansum[gr * 1024 + c] / fmaxf((float)cnt[gr], 1.f)
                       : maxp[gr * 1024 + (c - 1024)];
  g[i] = v;
  gbf[i] = (bf16)v;
}

// final classifier: out[64,5] = hcls[64,256] @ Wc2[256,5] + bc2
__global__ void final_kernel(const float* __restrict__ hcls, const float* __restrict__ Wc2,
                             const float* __restrict__ bc2, float* __restrict__ out, int G) {
  int i = blockIdx.x * blockDim.x + threadIdx.x;
  if (i >= G * 5) return;
  int g = i / 5, c = i % 5;
  float acc = bc2[c];
  for (int k = 0; k < 256; ++k) acc += hcls[g * 256 + k] * Wc2[k * 5 + c];
  out[i] = acc;
}

// ---------------------------------------------------------------------------
// Host orchestration
// ---------------------------------------------------------------------------
static inline unsigned nblk(size_t n, int tpb) { return (unsigned)((n + tpb - 1) / tpb); }

extern "C" void kernel_launch(void* const* d_in, const int* in_sizes, int n_in,
                              void* d_out, int out_size, void* d_ws, size_t ws_size,
                              hipStream_t stream) {
  (void)n_in; (void)out_size; (void)ws_size;
  const float* x   = (const float*)d_in[0];
  const int*   ei  = (const int*)d_in[1];
  const int*   bat = (const int*)d_in[2];
  const float* Wp  = (const float*)d_in[3];
  const float* bp  = (const float*)d_in[4];
  const float* W1  = (const float*)d_in[5];
  const float* as1 = (const float*)d_in[6];
  const float* ad1 = (const float*)d_in[7];
  const float* b1  = (const float*)d_in[8];
  const float* W2  = (const float*)d_in[9];
  const float* as2 = (const float*)d_in[10];
  const float* ad2 = (const float*)d_in[11];
  const float* b2  = (const float*)d_in[12];
  const float* Wc1 = (const float*)d_in[13];
  const float* bc1 = (const float*)d_in[14];
  const float* Wc2 = (const float*)d_in[15];
  const float* bc2 = (const float*)d_in[16];

  const int IN = 768, HD = 256, HC = 1024, G = 64;
  const int N  = in_sizes[0] / IN;
  const int E  = in_sizes[1] / 2;
  const int E2 = E + N;

  // workspace layout (256B aligned)
  char* ws = (char*)d_ws;
  size_t off = 0;
  auto take = [&](size_t bytes) -> char* {
    char* p = ws + off;
    off = (off + bytes + 255) & ~(size_t)255;
    return p;
  };
  bf16*  bfA     = (bf16*)take((size_t)N * HC * 2);        // x_bf / h0_bf / h1_bf
  bf16*  bfW     = (bf16*)take((size_t)1024 * 1024 * 2);   // current weight, bf16 TRANSPOSED
  float* P       = (float*)take((size_t)N * HC * 4);       // hW buffer
  float* Q       = (float*)take((size_t)N * HC * 4);       // aggregation / h buffer
  float* a_s     = (float*)take((size_t)N * 4 * 4);
  float* a_d     = (float*)take((size_t)N * 4 * 4);
  float* mbuf    = (float*)take((size_t)N * 4 * 4);
  float* dbuf    = (float*)take((size_t)N * 4 * 4);
  float* meansum = (float*)take((size_t)G * HC * 4);
  float* maxp    = (float*)take((size_t)G * HC * 4);
  int*   cnt     = (int*)  take((size_t)G * 4);
  float* gfeat   = (float*)take((size_t)G * 2 * HC * 4);
  bf16*  gbf     = (bf16*) take((size_t)G * 2 * HC * 2);
  float* hcls    = (float*)take((size_t)G * 256 * 4);

  bf16* x_bf  = bfA;
  bf16* h0_bf = bfA + (size_t)N * IN;  // disjoint from x_bf
  bf16* h1_bf = bfA;

  const int T = 256;

  // ---- input projection: h0 = relu(x @ Wp + bp) ----
  cvt_bf16_kernel<<<nblk((size_t)N * IN, T), T, 0, stream>>>(x, x_bf, (size_t)N * IN);
  cvt_tr_bf16_kernel<<<nblk((size_t)IN * HD, T), T, 0, stream>>>(Wp, bfW, IN, HD);
  wmma_gemm_bf16<<<dim3(HD / TNB, (N + TM - 1) / TM), T, 0, stream>>>(
      x_bf, bfW, bp, P, h0_bf, N, HD, IN, 1);

  // ---- GAT layer 1 ----
  cvt_tr_bf16_kernel<<<nblk((size_t)HD * HC, T), T, 0, stream>>>(W1, bfW, HD, HC);
  wmma_gemm_bf16<<<dim3(HC / TNB, (N + TM - 1) / TM), T, 0, stream>>>(
      h0_bf, bfW, nullptr, P, nullptr, N, HC, HD, 0);
  scores_kernel<<<nblk((size_t)N * 4 * 32, T), T, 0, stream>>>(P, as1, ad1, a_s, a_d, N);
  fill_f32_kernel<<<nblk((size_t)N * 4, T), T, 0, stream>>>(mbuf, -1e30f, (size_t)N * 4);
  fill_f32_kernel<<<nblk((size_t)N * 4, T), T, 0, stream>>>(dbuf, 0.f, (size_t)N * 4);
  fill_f32_kernel<<<nblk((size_t)N * HC, T), T, 0, stream>>>(Q, 0.f, (size_t)N * HC);
  edge_max_kernel<<<nblk((size_t)E2 * 4, T), T, 0, stream>>>(ei, E, N, a_s, a_d, mbuf);
  edge_sum_kernel<<<nblk((size_t)E2 * 4, T), T, 0, stream>>>(ei, E, N, a_s, a_d, mbuf, dbuf);
  edge_agg_kernel<<<nblk((size_t)E2 * 32, T), T, 0, stream>>>(ei, E, N, a_s, a_d, mbuf, dbuf, P, Q);
  bias_elu_kernel<<<nblk((size_t)N * HC, T), T, 0, stream>>>(Q, b1, h1_bf, (size_t)N * HC, HC);

  // ---- GAT layer 2 ----
  cvt_tr_bf16_kernel<<<nblk((size_t)HC * HC, T), T, 0, stream>>>(W2, bfW, HC, HC);
  wmma_gemm_bf16<<<dim3(HC / TNB, (N + TM - 1) / TM), T, 0, stream>>>(
      h1_bf, bfW, nullptr, P, nullptr, N, HC, HC, 0);
  scores_kernel<<<nblk((size_t)N * 4 * 32, T), T, 0, stream>>>(P, as2, ad2, a_s, a_d, N);
  fill_f32_kernel<<<nblk((size_t)N * 4, T), T, 0, stream>>>(mbuf, -1e30f, (size_t)N * 4);
  fill_f32_kernel<<<nblk((size_t)N * 4, T), T, 0, stream>>>(dbuf, 0.f, (size_t)N * 4);
  fill_f32_kernel<<<nblk((size_t)N * HC, T), T, 0, stream>>>(Q, 0.f, (size_t)N * HC);
  edge_max_kernel<<<nblk((size_t)E2 * 4, T), T, 0, stream>>>(ei, E, N, a_s, a_d, mbuf);
  edge_sum_kernel<<<nblk((size_t)E2 * 4, T), T, 0, stream>>>(ei, E, N, a_s, a_d, mbuf, dbuf);
  edge_agg_kernel<<<nblk((size_t)E2 * 32, T), T, 0, stream>>>(ei, E, N, a_s, a_d, mbuf, dbuf, P, Q);
  bias_elu_kernel<<<nblk((size_t)N * HC, T), T, 0, stream>>>(Q, b2, nullptr, (size_t)N * HC, HC);

  // ---- pooling ----
  fill_f32_kernel<<<nblk((size_t)G * HC, T), T, 0, stream>>>(meansum, 0.f, (size_t)G * HC);
  fill_f32_kernel<<<nblk((size_t)G * HC, T), T, 0, stream>>>(maxp, -1e30f, (size_t)G * HC);
  fill_f32_kernel<<<nblk((size_t)G, T), T, 0, stream>>>((float*)cnt, 0.f, (size_t)G);
  pool_accum_kernel<<<nblk((size_t)N * HC, T), T, 0, stream>>>(Q, bat, meansum, maxp, cnt, N);
  pool_finalize_kernel<<<nblk((size_t)G * 2 * HC, T), T, 0, stream>>>(meansum, maxp, cnt, gfeat, gbf, G);

  // ---- classifier ----
  cvt_tr_bf16_kernel<<<nblk((size_t)2 * HC * 256, T), T, 0, stream>>>(Wc1, bfW, 2 * HC, 256);
  wmma_gemm_bf16<<<dim3(256 / TNB, (G + TM - 1) / TM), T, 0, stream>>>(
      gbf, bfW, bc1, hcls, nullptr, G, 256, 2 * HC, 1);
  final_kernel<<<nblk((size_t)G * 5, T), T, 0, stream>>>(hcls, Wc2, bc2, (float*)d_out, G);
}
